// MPNNEncoder_11690900979945
// MI455X (gfx1250) — matrix operand order repaired and checked by reading, not compiled, not run on hardware
//
#include <hip/hip_runtime.h>
#include <hip/hip_bf16.h>

// Problem constants (match reference)
static constexpr int NN = 300000;   // nodes
static constexpr int EE = 600000;   // edges
static constexpr int GG = 10000;    // graphs
static constexpr int HH = 128;      // hidden
static constexpr int AF = 9;        // atom features
static constexpr int EF = 3;        // bond features
static constexpr int LL = 3;        // layers
#define BN_EPS 1e-5f

typedef __bf16 bf16_t;
typedef __attribute__((ext_vector_type(16))) __bf16 bf16x16;
typedef __attribute__((ext_vector_type(8)))  float  f32x8;
typedef __attribute__((ext_vector_type(4)))  float  f32x4;

// ---------------------------------------------------------------------------
// Small-K embedding: out[r][c] = sum_k X[r][k] * W[k][HH=128 cols] + B[c]
// ---------------------------------------------------------------------------
__global__ void embed_kernel(const float* __restrict__ X, const float* __restrict__ W,
                             const float* __restrict__ B, float* __restrict__ out,
                             long long rows, int K) {
  long long idx = (long long)blockIdx.x * blockDim.x + threadIdx.x;
  if (idx >= rows * HH) return;
  int c = (int)(idx & (HH - 1));
  long long n = idx >> 7;
  const float* xr = X + n * K;
  float acc = B[c];
  for (int k = 0; k < K; ++k) acc = fmaf(xr[k], W[k * HH + c], acc);
  out[idx] = acc;
}

// ---------------------------------------------------------------------------
// GINE message + scatter: agg[dst] += relu(h[src] + ea)
// One thread per (edge, 4-channel group): float4 gathers + 4 f32 atomics.
// ---------------------------------------------------------------------------
__global__ void msg_kernel(const float* __restrict__ h, const float* __restrict__ ea,
                           const long long* __restrict__ src, const long long* __restrict__ dst,
                           float* __restrict__ agg) {
  long long idx = (long long)blockIdx.x * blockDim.x + threadIdx.x;
  if (idx >= (long long)EE * (HH / 4)) return;
  int cg = (int)(idx & 31);          // 32 groups of 4 channels
  long long e = idx >> 5;
  long long s = src[e];
  long long d = dst[e];
  const f32x4 hv = *reinterpret_cast<const f32x4*>(h + s * HH + cg * 4);
  const f32x4 ev = *reinterpret_cast<const f32x4*>(ea + e * HH + cg * 4);
  float* ag = agg + d * HH + cg * 4;
#pragma unroll
  for (int j = 0; j < 4; ++j)
    atomicAdd(&ag[j], fmaxf(hv[j] + ev[j], 0.0f));
}

// ---------------------------------------------------------------------------
// WMMA GEMM: C[M x 128] = f(A[M x 128]) @ W[128 x 128] + bias
//   mode 0: f = identity
//   mode 1: f(x) = relu(x * scale[k] + shift[k])   (fused BN+ReLU on input)
// Optionally accumulates per-column sum / sumsq of C into stats[2][128].
// 256 threads = 8 waves; wave w covers rows [blk*128 + 16w, +16) x all 128 cols.
// A loads: unconditional vectorized b128 from a clamped-safe row; invalid rows
// are neutralized by a mask multiply (keeps EXEC all-ones for WMMA, no branches).
// ---------------------------------------------------------------------------
__global__ __launch_bounds__(256)
void gemm128_wmma(const float* __restrict__ A, const float* __restrict__ W,
                  const float* __restrict__ bias, float* __restrict__ C,
                  long long M, int mode,
                  const float* __restrict__ scale, const float* __restrict__ shift,
                  float* __restrict__ stats)
{
  // Weights transposed into LDS: Bs[n][k] = W[k][n]; pad row to 144 bf16 (288 B)
  __shared__ __align__(32) bf16_t Bs[HH][HH + 16];
  const int tid = threadIdx.x;
  for (int i = tid; i < HH * HH; i += 256) {
    int k = i >> 7, n = i & (HH - 1);
    Bs[n][k] = (bf16_t)W[i];
  }
  __syncthreads();

  const int wave  = tid >> 5;
  const int lane  = tid & 31;
  const int half  = lane >> 4;   // lane group 0..1
  const int mlane = lane & 15;

  // A-fragment row for this lane (16-bit A layout: both lane halves M=0..15)
  const long long rowA = (long long)blockIdx.x * 128 + wave * 16 + mlane;
  const bool inA = rowA < M;
  const long long rowSafe = inA ? rowA : 0;          // safe address for dead lanes
  const float msk = inA ? 1.0f : 0.0f;
  const f32x4* __restrict__ arow4 = reinterpret_cast<const f32x4*>(A + rowSafe * HH);
  const f32x4* __restrict__ scl4  = reinterpret_cast<const f32x4*>(scale);
  const f32x4* __restrict__ shf4  = reinterpret_cast<const f32x4*>(shift);

  f32x8 acc[8];
#pragma unroll
  for (int t = 0; t < 8; ++t)
#pragma unroll
    for (int r = 0; r < 8; ++r) acc[t][r] = 0.0f;

#pragma unroll
  for (int ks = 0; ks < 4; ++ks) {            // K = 128 in 4 steps of 32
    const int kbase = ks * 32;
    // A K order used here: elem e -> K = kbase + 16*half + e (contiguous, matches B)
    const int b0 = (kbase + half * 16) >> 2;  // float4 index of first element
    f32x4 p[4];
    p[0] = arow4[b0];
    p[1] = arow4[b0 + 1];
    p[2] = arow4[b0 + 2];
    p[3] = arow4[b0 + 3];
    if (mode == 1) {
#pragma unroll
      for (int q = 0; q < 4; ++q) {
        const int sb = b0 + q;   // matches p offsets
        const f32x4 sc = scl4[sb];
        const f32x4 sh = shf4[sb];
#pragma unroll
        for (int j = 0; j < 4; ++j)
          p[q][j] = fmaxf(fmaf(p[q][j], sc[j], sh[j]), 0.0f);
      }
    }
    bf16x16 a;
#pragma unroll
    for (int q = 0; q < 4; ++q)
#pragma unroll
      for (int j = 0; j < 4; ++j)
        a[q * 4 + j] = (bf16_t)(p[q][j] * msk);

#pragma unroll
    for (int nt = 0; nt < 8; ++nt) {
      const int n = nt * 16 + mlane;
      // B K order used here (same as A): elem e -> K = kbase + 16*half + e, N = mlane
      bf16x16 b = *reinterpret_cast<const bf16x16*>(&Bs[n][kbase + half * 16]);
      acc[nt] = __builtin_amdgcn_wmma_f32_16x16x32_bf16(
          false, a, false, b, (short)0, acc[nt], false, false);
    }
  }

  // C layout: v8f elem r -> M = r + 8*half, N = mlane (per 16x16 tile)
  const long long rowC0 = (long long)blockIdx.x * 128 + wave * 16 + half * 8;
  const bool full = (rowC0 + 8) <= M;
#pragma unroll
  for (int nt = 0; nt < 8; ++nt) {
    const int n = nt * 16 + mlane;
    const float bv = bias[n];
    float s = 0.0f, s2 = 0.0f;
    if (full) {
#pragma unroll
      for (int r = 0; r < 8; ++r) {
        float v = acc[nt][r] + bv;
        C[(rowC0 + r) * HH + n] = v;
        s += v;
        s2 += v * v;
      }
    } else {
#pragma unroll
      for (int r = 0; r < 8; ++r) {
        if (rowC0 + r < M) {
          float v = acc[nt][r] + bv;
          C[(rowC0 + r) * HH + n] = v;
          s += v;
          s2 += v * v;
        }
      }
    }
    if (stats != nullptr) {
      atomicAdd(&stats[n], s);
      atomicAdd(&stats[HH + n], s2);
    }
  }
}

// ---------------------------------------------------------------------------
// Fold BN batch stats + affine into per-channel scale/shift:
//   bn(x) = x*scale + shift, scale = g*rsqrt(var+eps), shift = be - mean*scale
// ---------------------------------------------------------------------------
__global__ void bn_finalize_kernel(const float* __restrict__ stats, float invM,
                                   const float* __restrict__ g, const float* __restrict__ be,
                                   float* __restrict__ scale, float* __restrict__ shift) {
  int c = threadIdx.x;
  float mean = stats[c] * invM;
  float var  = stats[HH + c] * invM - mean * mean;
  float rs   = rsqrtf(var + BN_EPS);
  float sc   = g[c] * rs;
  scale[c] = sc;
  shift[c] = be[c] - mean * sc;
}

// h = relu(z * scale + shift)
__global__ void update_h_kernel(const float* __restrict__ z,
                                const float* __restrict__ scale,
                                const float* __restrict__ shift,
                                float* __restrict__ h, long long total) {
  long long idx = (long long)blockIdx.x * blockDim.x + threadIdx.x;
  if (idx >= total) return;
  int c = (int)(idx & (HH - 1));
  h[idx] = fmaxf(fmaf(z[idx], scale[c], shift[c]), 0.0f);
}

__global__ void zero_kernel(float* __restrict__ p, long long n) {
  long long idx = (long long)blockIdx.x * blockDim.x + threadIdx.x;
  if (idx < n) p[idx] = 0.0f;
}

// Global mean pool (sum + count via atomics); float4 per thread
__global__ void pool_kernel(const float* __restrict__ h, const long long* __restrict__ batch,
                            float* __restrict__ pooled, float* __restrict__ cnt) {
  long long idx = (long long)blockIdx.x * blockDim.x + threadIdx.x;
  if (idx >= (long long)NN * (HH / 4)) return;
  int cg = (int)(idx & 31);
  long long n = idx >> 5;
  long long g = batch[n];
  const f32x4 hv = *reinterpret_cast<const f32x4*>(h + n * HH + cg * 4);
  float* pg = pooled + g * HH + cg * 4;
#pragma unroll
  for (int j = 0; j < 4; ++j) atomicAdd(&pg[j], hv[j]);
  if (cg == 0) atomicAdd(&cnt[g], 1.0f);
}

__global__ void pool_norm_kernel(const float* __restrict__ pooled, const float* __restrict__ cnt,
                                 float* __restrict__ out) {
  long long idx = (long long)blockIdx.x * blockDim.x + threadIdx.x;
  if (idx >= (long long)GG * HH) return;
  long long g = idx >> 7;
  out[idx] = pooled[idx] / fmaxf(cnt[g], 1.0f);
}

// ---------------------------------------------------------------------------
extern "C" void kernel_launch(void* const* d_in, const int* in_sizes, int n_in,
                              void* d_out, int out_size, void* d_ws, size_t ws_size,
                              hipStream_t stream) {
  (void)in_sizes; (void)n_in; (void)out_size; (void)ws_size;
  const float*     x      = (const float*)d_in[0];
  const long long* ei     = (const long long*)d_in[1];   // int64 [2, E]
  const float*     eattr  = (const float*)d_in[2];
  const long long* batch  = (const long long*)d_in[3];   // int64 [N]
  const float* w_atom = (const float*)d_in[4];
  const float* b_atom = (const float*)d_in[5];
  const float* w_bond = (const float*)d_in[6];
  const float* b_bond = (const float*)d_in[7];
  const float* w1     = (const float*)d_in[8];
  const float* b1     = (const float*)d_in[9];
  const float* g1     = (const float*)d_in[10];
  const float* be1    = (const float*)d_in[11];
  const float* w2     = (const float*)d_in[12];
  const float* b2     = (const float*)d_in[13];
  const float* g_out  = (const float*)d_in[14];
  const float* be_out = (const float*)d_in[15];
  const float* w_proj = (const float*)d_in[16];
  const float* b_proj = (const float*)d_in[17];
  float* out = (float*)d_out;

  // Workspace partition (256 B aligned)
  char* wsp = (char*)d_ws;
  auto alloc = [&](size_t bytes) -> float* {
    char* p = wsp;
    wsp += (bytes + 255) & ~(size_t)255;
    return (float*)p;
  };
  float* h    = alloc((size_t)NN * HH * 4);
  float* ea   = alloc((size_t)EE * HH * 4);
  float* agg  = alloc((size_t)NN * HH * 4);   // doubles as z2 after gemm2
  float* z1   = alloc((size_t)NN * HH * 4);
  float* stats= alloc(2 * HH * 4);
  float* sc1  = alloc(HH * 4);
  float* sh1  = alloc(HH * 4);
  float* sc2  = alloc(HH * 4);
  float* sh2  = alloc(HH * 4);
  float* pooled  = alloc((size_t)GG * HH * 4);
  float* cnt     = alloc((size_t)GG * 4);
  float* pooledN = alloc((size_t)GG * HH * 4);

  const long long NHtot = (long long)NN * HH;
  const long long EHtot = (long long)EE * HH;
  dim3 blk(256);
  auto nb = [](long long total) { return dim3((unsigned)((total + 255) / 256)); };

  // Embeddings
  embed_kernel<<<nb(NHtot), blk, 0, stream>>>(x, w_atom, b_atom, h, NN, AF);
  embed_kernel<<<nb(EHtot), blk, 0, stream>>>(eattr, w_bond, b_bond, ea, EE, EF);

  const long long* src = ei;        // edge_index[0]
  const long long* dst = ei + EE;   // edge_index[1]
  const unsigned gemmBlocksN = (NN + 127) / 128;

  for (int l = 0; l < LL; ++l) {
    const float* w1l = w1 + (size_t)l * HH * HH;
    const float* b1l = b1 + l * HH;
    const float* g1l = g1 + l * HH;
    const float* be1l = be1 + l * HH;
    const float* w2l = w2 + (size_t)l * HH * HH;
    const float* b2l = b2 + l * HH;
    const float* gol = g_out + l * HH;
    const float* bol = be_out + l * HH;

    // agg = h, then scatter-add messages
    hipMemcpyAsync(agg, h, (size_t)NN * HH * 4, hipMemcpyDeviceToDevice, stream);
    msg_kernel<<<nb((long long)EE * 32), blk, 0, stream>>>(h, ea, src, dst, agg);

    // z1 = agg @ w1 + b1 (+ stats for BN1)
    zero_kernel<<<1, 256, 0, stream>>>(stats, 2 * HH);
    gemm128_wmma<<<gemmBlocksN, blk, 0, stream>>>(agg, w1l, b1l, z1, NN, 0,
                                                  nullptr, nullptr, stats);
    bn_finalize_kernel<<<1, HH, 0, stream>>>(stats, 1.0f / NN, g1l, be1l, sc1, sh1);

    // z2 = relu(bn1(z1)) @ w2 + b2 (+ stats for outer BN); z2 reuses agg buffer
    zero_kernel<<<1, 256, 0, stream>>>(stats, 2 * HH);
    gemm128_wmma<<<gemmBlocksN, blk, 0, stream>>>(z1, w2l, b2l, agg, NN, 1,
                                                  sc1, sh1, stats);
    bn_finalize_kernel<<<1, HH, 0, stream>>>(stats, 1.0f / NN, gol, bol, sc2, sh2);

    // h = relu(bn_out(z2))
    update_h_kernel<<<nb(NHtot), blk, 0, stream>>>(agg, sc2, sh2, h, NHtot);
  }

  // Global mean pool
  zero_kernel<<<nb((long long)GG * HH), blk, 0, stream>>>(pooled, (long long)GG * HH);
  zero_kernel<<<nb(GG), blk, 0, stream>>>(cnt, GG);
  pool_kernel<<<nb((long long)NN * 32), blk, 0, stream>>>(h, batch, pooled, cnt);
  pool_norm_kernel<<<nb((long long)GG * HH), blk, 0, stream>>>(pooled, cnt, pooledN);

  // Final projection: out = pooledN @ w_proj + b_proj
  gemm128_wmma<<<(GG + 127) / 128, blk, 0, stream>>>(pooledN, w_proj, b_proj, out,
                                                     GG, 0, nullptr, nullptr, nullptr);
}
